// CharRNN_86792699118069
// MI455X (gfx1250) — compile-verified, matrix operand order checked by reading, not run on hardware
//
#include <hip/hip_runtime.h>
#include <math.h>

#define VOCAB  33
#define EMBED  200
#define HIDDEN 200
#define BATCH  256
#define SEQ    1024

#define NPAD 208   // HIDDEN padded to 13*16 (N tiles)
#define KPAD 224   // HIDDEN padded to 7*32  (K tiles)
#define NT   13    // N tiles over hidden
#define KT   7     // K tiles over hidden
#define VT   3     // N tiles over vocab (48)

#define ABUF_ELEMS (KT * 512)   // 3584 bf16 per buffer

typedef __bf16 bf16;
typedef __attribute__((ext_vector_type(16))) __bf16 v16bf;
typedef __attribute__((ext_vector_type(8)))  float  v8f;

// Branchless tanh: native v_tanh_f32 on gfx1250 if the builtin exists,
// else 1 - 2/(exp(2x)+1) via native v_exp_f32 / v_rcp_f32 (exact at +-inf).
__device__ __forceinline__ float tanh_fast(float v) {
#if __has_builtin(__builtin_amdgcn_tanhf)
    return __builtin_amdgcn_tanhf(v);
#else
    float e = __expf(2.f * v);
    return 1.f - 2.f * __builtin_amdgcn_rcpf(e + 1.f);
#endif
}

// ---------------------------------------------------------------------------
// Prep 1: EWT = (embedding @ W_e)^T, stored [NPAD][VOCAB] (pad rows = 0).
// embedding[x] @ W_e == (embedding @ W_e)[x]: the 21 GFLOP input projection
// collapses to a 33x200x200 GEMM + lookups. Transposed layout puts one hidden
// column's 33 token values in a 132 B row -> a lane's 8 gathers share lines.
// ---------------------------------------------------------------------------
__global__ void prep_ew(const float* __restrict__ emb, const float* __restrict__ We,
                        float* __restrict__ EWT) {
    int idx = blockIdx.x * blockDim.x + threadIdx.x;
    if (idx >= NPAD * VOCAB) return;
    int j = idx / VOCAB, i = idx % VOCAB;      // j = hidden col, i = token
    float s = 0.f;
    if (j < HIDDEN) {
        for (int e = 0; e < EMBED; ++e) s += emb[i * EMBED + e] * We[e * HIDDEN + j];
    }
    EWT[idx] = s;
}

// ---------------------------------------------------------------------------
// Prep 2/3: pack W_h / W_o into v_wmma_f32_16x16x32_bf16 B-fragment layout
// [ntile][ktile][lane][16 bf16], zero-padded (ISA 7.12.2, 16-bit, wave32):
//   klocal = (e>=8 ? 16:0) + (lane>=16 ? 8:0) + (e&7),  n = 16*tile + (lane&15)
// ---------------------------------------------------------------------------
__global__ void pack_wh(const float* __restrict__ Wh, bf16* __restrict__ frag) {
    int idx = blockIdx.x * blockDim.x + threadIdx.x;
    if (idx >= NT * KT * 512) return;
    int e = idx & 15, lane = (idx >> 4) & 31, rem = idx >> 9;
    int kt = rem % KT, w = rem / KT;
    int kl = ((e >= 8) ? 16 : 0) + ((lane >= 16) ? 8 : 0) + (e & 7);
    int k = kt * 32 + kl;
    int n = w * 16 + (lane & 15);
    float v = (k < HIDDEN && n < HIDDEN) ? Wh[k * HIDDEN + n] : 0.f;
    frag[idx] = (bf16)v;
}

__global__ void pack_wo(const float* __restrict__ Wo, bf16* __restrict__ frag) {
    int idx = blockIdx.x * blockDim.x + threadIdx.x;
    if (idx >= VT * KT * 512) return;
    int e = idx & 15, lane = (idx >> 4) & 31, rem = idx >> 9;
    int kt = rem % KT, vt = rem / KT;
    int kl = ((e >= 8) ? 16 : 0) + ((lane >= 16) ? 8 : 0) + (e & 7);
    int k = kt * 32 + kl;
    int n = vt * 16 + (lane & 15);
    float v = (k < HIDDEN && n < VOCAB) ? Wo[k * VOCAB + n] : 0.f;
    frag[idx] = (bf16)v;
}

// ---------------------------------------------------------------------------
// Persistent scan: 16 workgroups (one per 16-row batch strip), 16 waves each.
// Waves 0..12: recurrence N-tiles. Waves 13..15: fused logits GEMM one step
// behind (non-temporal stores to HBM). h ping-pongs in LDS in exact A-fragment
// layout; W_h / W_o fragments live in VGPRs for all 1024 steps.
// Per step per wave: issue 8 EW gathers -> 14 ds_load_b128 -> 7 v_wmma
// (gathers land under the WMMAs) -> 8 branchless tanh -> bf16 LDS scatter
// (precomputed offsets) -> 1 split barrier. No branches on the critical path.
// ---------------------------------------------------------------------------
__global__ void __launch_bounds__(512)
rnn_scan(const int* __restrict__ x, const float* __restrict__ hidden,
         const float* __restrict__ EWT, const bf16* __restrict__ Whf,
         const bf16* __restrict__ Wof, float* __restrict__ out) {
    __shared__ __align__(32) bf16 Abuf[2][ABUF_ELEMS];
    __shared__ __align__(16) int tok[2][16];

    const int strip = blockIdx.x * 16;
    const int tid   = threadIdx.x;
    const int wv    = tid >> 5;
    const int lane  = tid & 31;
    const int hif   = (lane >= 16) ? 1 : 0;
    const int nl    = lane & 15;

    float* logits = out;
    float* fhid   = out + (size_t)BATCH * SEQ * VOCAB;

    // Zero both A buffers: K-pad region [208,224) must stay exactly 0
    // (padded WMMA inputs then contribute 0, never NaN).
    for (int i = tid; i < 2 * ABUF_ELEMS; i += 512) (&Abuf[0][0])[i] = (bf16)0.f;
    __syncthreads();

    // Scatter initial hidden state into A-fragment layout of buffer 0.
    for (int i = tid; i < 16 * HIDDEN; i += 512) {
        int m = i / HIDDEN, k = i % HIDDEN;
        int kt = k >> 5, kl = k & 31;
        int e  = ((kl & 16) ? 8 : 0) + (kl & 7);
        int la = m + ((kl & 8) ? 16 : 0);
        Abuf[0][(kt * 32 + la) * 16 + e] = (bf16)hidden[(strip + m) * HIDDEN + k];
    }
    if (tid < 16) tok[0][tid] = x[(strip + tid) * SEQ + 0];

    // Weight fragments -> registers (invariant across all 1024 steps).
    v16bf bfr[KT];
    if (wv < NT) {
        #pragma unroll
        for (int kt = 0; kt < KT; ++kt)
            bfr[kt] = *(const v16bf*)(Whf + ((size_t)(wv * KT + kt) * 32 + lane) * 16);
    } else {
        int vt = wv - NT;
        #pragma unroll
        for (int kt = 0; kt < KT; ++kt)
            bfr[kt] = *(const v16bf*)(Wof + ((size_t)(vt * KT + kt) * 32 + lane) * 16);
    }

    // Loop-invariant pieces for the recurrence gather + scatter.
    const int n_col = wv * 16 + nl;              // hidden column (waves 0..12)
    const float* ewrow = EWT + n_col * VOCAB;    // gather base: ewrow[token]
    int sc_off[8];                               // LDS scatter offsets for k = n_col
    {
        int kl = n_col & 31;
        int e  = ((kl & 16) ? 8 : 0) + (kl & 7);
        int kb = (n_col >> 5) * 32;
        int lb = ((kl & 8) ? 16 : 0);
        #pragma unroll
        for (int r = 0; r < 8; ++r)
            sc_off[r] = (kb + (r + 8 * hif) + lb) * 16 + e;
    }
    const int v_col = (wv - NT) * 16 + nl;       // vocab column (waves 13..15)
    const int aload = lane * 16;                 // A-fragment load base (elems)

    __syncthreads();

    bf16* Ac = &Abuf[0][0];                      // holds h_{t-1}
    bf16* An = &Abuf[1][0];                      // receives h_t
    for (int t = 0; t < SEQ; ++t) {
        // Token prefetch for t+1, done by wave 15 (least-loaded logits wave).
        if (wv == 15 && lane < 16 && t + 1 < SEQ)
            tok[(t + 1) & 1][lane] = x[(strip + lane) * SEQ + t + 1];

        // Issue this step's 8 input-projection gathers FIRST so they complete
        // in the shadow of the WMMA pipeline below.
        float xw[8];
        if (wv < NT) {
            const int4 ta = *(const int4*)&tok[t & 1][8 * hif];
            const int4 tb = *(const int4*)&tok[t & 1][8 * hif + 4];
            const int tk[8] = {ta.x, ta.y, ta.z, ta.w, tb.x, tb.y, tb.z, tb.w};
            #pragma unroll
            for (int r = 0; r < 8; ++r) xw[r] = ewrow[tk[r]];
        }

        // A (= h_{t-1}) from LDS, 7 chained bf16 WMMAs with f32 accumulate.
        v8f acc = {};
        #pragma unroll
        for (int kt = 0; kt < KT; ++kt) {
            v16bf a = *(const v16bf*)(Ac + kt * 512 + aload);
            acc = __builtin_amdgcn_wmma_f32_16x16x32_bf16(
                false, a, false, bfr[kt], (short)0, acc, false, false);
        }

        if (wv < NT) {
            // h_t = tanh(xW + h_{t-1} @ W_h); branchless tanh, scatter to LDS.
            #pragma unroll
            for (int r = 0; r < 8; ++r) {
                float h = tanh_fast(xw[r] + acc[r]);
                An[sc_off[r]] = (bf16)h;
            }
        } else if (t > 0 && v_col < VOCAB) {
            // logits_{t-1} = h_{t-1} @ W_o, streamed (non-temporal) to HBM.
            #pragma unroll
            for (int r = 0; r < 8; ++r) {
                int m = r + 8 * hif;
                __builtin_nontemporal_store(
                    acc[r], &logits[((size_t)(strip + m) * SEQ + (t - 1)) * VOCAB + v_col]);
            }
        }
        __syncthreads();
        bf16* tmp = Ac; Ac = An; An = tmp;       // Ac now holds h_t
    }

    // Tail 1: logits for the final step from Ac (= h_{SEQ-1}).
    if (wv >= NT && v_col < VOCAB) {
        v8f acc = {};
        #pragma unroll
        for (int kt = 0; kt < KT; ++kt) {
            v16bf a = *(const v16bf*)(Ac + kt * 512 + aload);
            acc = __builtin_amdgcn_wmma_f32_16x16x32_bf16(
                false, a, false, bfr[kt], (short)0, acc, false, false);
        }
        #pragma unroll
        for (int r = 0; r < 8; ++r) {
            int m = r + 8 * hif;
            __builtin_nontemporal_store(
                acc[r], &logits[((size_t)(strip + m) * SEQ + (SEQ - 1)) * VOCAB + v_col]);
        }
    }

    // Tail 2: final hidden state h_{SEQ-1} (read back from A-fragment layout).
    for (int i = tid; i < 16 * HIDDEN; i += 512) {
        int m = i / HIDDEN, k = i % HIDDEN;
        int kt = k >> 5, kl = k & 31;
        int e  = ((kl & 16) ? 8 : 0) + (kl & 7);
        int la = m + ((kl & 8) ? 16 : 0);
        fhid[(size_t)(strip + m) * HIDDEN + k] = (float)Ac[(kt * 32 + la) * 16 + e];
    }
}

// ---------------------------------------------------------------------------
extern "C" void kernel_launch(void* const* d_in, const int* in_sizes, int n_in,
                              void* d_out, int out_size, void* d_ws, size_t ws_size,
                              hipStream_t stream) {
    const int*   xin = (const int*)d_in[0];
    const float* hid = (const float*)d_in[1];
    const float* emb = (const float*)d_in[2];
    const float* We  = (const float*)d_in[3];
    const float* Wh  = (const float*)d_in[4];
    const float* Wo  = (const float*)d_in[5];
    float* out = (float*)d_out;

    char* ws = (char*)d_ws;
    float* EWT = (float*)ws;                                 // 208*33*4  = 27456 B
    bf16*  Whf = (bf16*)(ws + 27648);                        // 13*7*512*2 = 93184 B
    bf16*  Wof = (bf16*)(ws + 27648 + 93184);                // 3*7*512*2  = 21504 B

    prep_ew<<<(NPAD * VOCAB + 255) / 256, 256, 0, stream>>>(emb, We, EWT);
    pack_wh<<<(NT * KT * 512 + 255) / 256, 256, 0, stream>>>(Wh, Whf);
    pack_wo<<<(VT * KT * 512 + 255) / 256, 256, 0, stream>>>(Wo, Wof);
    rnn_scan<<<BATCH / 16, 512, 0, stream>>>(xin, hid, EWT, Whf, Wof, out);
}
